// PickNMSPredictionsAndReturnAsBatchedResult_44744969290053
// MI455X (gfx1250) — compile-verified, matrix operand order checked by reading, not run on hardware
//
#include <hip/hip_runtime.h>
#include <hip/hip_bf16.h>

// ---------------------------------------------------------------------------
// PickNMSPredictionsAndReturnAsBatchedResult for MI455X (gfx1250, wave32)
//
// Sizes: B=64, N=8192, C=91, MAX_PRED=300, NUM_SEL=8000
// Outputs (flat, concatenated):
//   [0,        64)                       num_predictions  int32 (B,1)
//   [64,       64+76800)                 out_boxes        f32   (B,300,4)
//   [64+76800, 64+76800+19200)           out_scores       f32   (B,300)
//   [64+96000, 64+96000+19200)           out_classes      int32 (B,300)
// total = 115264 elements
// ---------------------------------------------------------------------------

#define BQ 64
#define NQ 8192
#define CQ 91
#define MAXP 300
#define NSEL 8000
#define OFF_BOX 64
#define OFF_SCR (64 + BQ * MAXP * 4)
#define OFF_CLS (64 + BQ * MAXP * 4 + BQ * MAXP)
#define OUT_TOT (64 + BQ * MAXP * 4 + BQ * MAXP + BQ * MAXP)

typedef int v8i __attribute__((ext_vector_type(8)));

// ---------------------------------------------------------------------------
// CDNA5 async global->LDS gather (per-lane B32, tracked by ASYNCcnt).
// Generic LDS pointers carry the wave-relative DS offset in their low 32 bits
// (ISA 10.2, LDS aperture), so we can pass the
// truncated generic address as the LDS destination VGPR.
// ---------------------------------------------------------------------------
__device__ __forceinline__ void async_gather_b32_to_lds(void* lds_dst,
                                                        const int* gsrc) {
    const unsigned lds_off = (unsigned)(size_t)lds_dst;
    asm volatile("global_load_async_to_lds_b32 %0, %1, off"
                 :
                 : "v"(lds_off), "v"(gsrc)
                 : "memory");
}

__device__ __forceinline__ void wait_asynccnt0() {
#if __has_builtin(__builtin_amdgcn_s_wait_asynccnt)
    __builtin_amdgcn_s_wait_asynccnt(0);
#else
    asm volatile("s_wait_asynccnt 0x0" ::: "memory");
#endif
}

// ---------------------------------------------------------------------------
// Kernel 0: zero the output buffer (reference output is zero-padded).
// ---------------------------------------------------------------------------
__global__ void zero_kernel(float* __restrict__ out, int n) {
    int i = blockIdx.x * blockDim.x + threadIdx.x;
    if (i < n) out[i] = 0.0f;
}

// ---------------------------------------------------------------------------
// Kernel 1: num_predictions via WMMA.
// segment_sum(ones, b) == onehot(b)^T * ones  -> a 16x16x64 IU8 matmul chain.
// 4 waves (128 threads); wave g owns batches [16g, 16g+16).
// A[m,k] = (batch[chunk*64+k] == m+16g) as u8, B = all-ones u8.
// D[m,n] += sum_k A[m,k]  (same value in every column n) = per-chunk histogram.
// 125 chunks of 64 cover all 8000 selections.
// ---------------------------------------------------------------------------
__global__ __launch_bounds__(128) void count_wmma_kernel(
    const int* __restrict__ sel,   // (NSEL,3) int32: [b, lab, box]
    float* __restrict__ out) {
    __shared__ __align__(16) int sbat[8192];

    const int tid = threadIdx.x;
    // Stage batch ids into LDS via async gather; sentinel 255 in the tail.
    for (int j = 0; j < 64; ++j) {
        int i = tid + j * 128;
        if (i < NSEL) {
            async_gather_b32_to_lds(&sbat[i], sel + 3 * i);
        } else {
            sbat[i] = 255;
        }
    }
    wait_asynccnt0();
    __syncthreads();

    const int lane = tid & 31;
    const int wav  = tid >> 5;          // 0..3 -> batch group
    const int m    = (lane & 15) + wav * 16;
    const int koff = (lane >= 16) ? 8 : 0;   // 8-bit A-matrix lane split

    v8i acc   = {0, 0, 0, 0, 0, 0, 0, 0};
    v8i bones = {0x01010101, 0x01010101, 0x01010101, 0x01010101,
                 0x01010101, 0x01010101, 0x01010101, 0x01010101};

    for (int c = 0; c < 125; ++c) {
        const int base = c * 64;
        v8i A;
#pragma unroll
        for (int v = 0; v < 8; ++v) {
            // K byte mapping for 16x64 u8 A (ISA 7.12.2):
            // vgpr bit0 -> +4, bit1 -> +16, bit2 -> +32; lanes>=16 -> +8.
            const int k0 = base + koff + (v & 1) * 4 + ((v >> 1) & 1) * 16 +
                           ((v >> 2) & 1) * 32;
            const int4 q = *(const int4*)&sbat[k0];
            int a = (q.x == m ? 1 : 0) | (q.y == m ? (1 << 8) : 0) |
                    (q.z == m ? (1 << 16) : 0) | (q.w == m ? (1 << 24) : 0);
            A[v] = a;
        }
        acc = __builtin_amdgcn_wmma_i32_16x16x64_iu8(
            /*sgn_a=*/false, A, /*sgn_b=*/false, bones, acc,
            /*reuse_a=*/false, /*reuse_b=*/false);
    }

    // 32-bit C/D layout: VGPR r holds M=r (lanes 0-15) and M=r+8 (lanes 16-31).
    int* np = (int*)out;
    if (lane == 0) {
#pragma unroll
        for (int r = 0; r < 8; ++r) np[wav * 16 + r] = acc[r];
    } else if (lane == 16) {
#pragma unroll
        for (int r = 0; r < 8; ++r) np[wav * 16 + 8 + r] = acc[r];
    }
}

// ---------------------------------------------------------------------------
// Kernel 2: stable segmented ranks + gather + scatter. One workgroup
// (1024 threads = 32 wave32s); index list lives entirely in LDS.
//  Phase 1: per-wave segment of 256 entries; local rank via wave32
//           ballot-based match_any + running per-wave 64-bin histogram in LDS.
//  Phase 2: exclusive scan of the 32 wave histograms per batch.
//  Phase 3: final rank = wave_base[b] + local; gather box/score, scatter row.
// ---------------------------------------------------------------------------
__global__ __launch_bounds__(1024) void rank_scatter_kernel(
    const float* __restrict__ pred_boxes,   // (B,N,4)
    const float* __restrict__ pred_scores,  // (B,N,C)
    const int* __restrict__ sel,            // (NSEL,3)
    float* __restrict__ out) {
    __shared__ __align__(16) int sbat[8192];
    __shared__ int whist[32 * 64];  // per-wave hist -> per-wave exclusive base

    const int tid  = threadIdx.x;
    const int lane = tid & 31;
    const int wav  = tid >> 5;

    // Stage batch ids via async gather; sentinel in the tail.
    for (int j = 0; j < 8; ++j) {
        int i = tid + j * 1024;
        if (i < NSEL) {
            async_gather_b32_to_lds(&sbat[i], sel + 3 * i);
        } else {
            sbat[i] = 255;
        }
    }
    for (int j = tid; j < 32 * 64; j += 1024) whist[j] = 0;
    wait_asynccnt0();
    __syncthreads();

    // Phase 1: each wave scans its 8 chunks of 32 entries.
    int vv[8];
    int lr[8];
    const unsigned ltmask = (1u << lane) - 1u;
#pragma unroll
    for (int t = 0; t < 8; ++t) {
        const int i     = wav * 256 + t * 32 + lane;
        const int valid = (i < NSEL) ? 1 : 0;
        const int v     = valid ? sbat[i] : 0;

        unsigned mask = __builtin_amdgcn_ballot_w32(valid != 0);
#pragma unroll
        for (int bit = 0; bit < 6; ++bit) {
            unsigned bb = __builtin_amdgcn_ballot_w32(((v >> bit) & 1) != 0);
            mask &= ((v >> bit) & 1) ? bb : ~bb;
        }
        const int before = __builtin_popcount(mask & ltmask);
        const int cur    = whist[wav * 64 + v];  // all lanes read first
        vv[t] = v;
        lr[t] = cur + before;
        const unsigned lead = mask & (~mask + 1u);  // lowest matching lane
        if (valid && lead == (1u << lane)) {
            whist[wav * 64 + v] = cur + __builtin_popcount(mask);
        }
    }
    __syncthreads();

    // Phase 2: exclusive scan over the 32 waves, one thread per batch.
    if (tid < BQ) {
        int run = 0;
        for (int w = 0; w < 32; ++w) {
            const int h = whist[w * 64 + tid];
            whist[w * 64 + tid] = run;  // in-place: becomes exclusive base
            run += h;
        }
    }
    __syncthreads();

    // Phase 3: gather + scatter.
#pragma unroll
    for (int t = 0; t < 8; ++t) {
        const int i = wav * 256 + t * 32 + lane;
        if (i >= NSEL) continue;
        const int b    = vv[t];
        const int rank = whist[wav * 64 + b] + lr[t];
        if (rank >= MAXP) continue;

        const int lab = sel[3 * i + 1];
        const int box = sel[3 * i + 2];
        const size_t bi = (size_t)b * NQ + (size_t)box;

        const float4 bx = *(const float4*)(pred_boxes + bi * 4);
        const float  sc = pred_scores[bi * CQ + (size_t)lab];

        const int slot = b * MAXP + rank;
        *(float4*)(out + OFF_BOX + (size_t)slot * 4) = bx;
        out[OFF_SCR + slot]                          = sc;
        ((int*)out)[OFF_CLS + slot]                  = lab;
    }
}

// ---------------------------------------------------------------------------
extern "C" void kernel_launch(void* const* d_in, const int* in_sizes, int n_in,
                              void* d_out, int out_size, void* d_ws,
                              size_t ws_size, hipStream_t stream) {
    const float* pred_boxes  = (const float*)d_in[0];
    const float* pred_scores = (const float*)d_in[1];
    const int*   sel         = (const int*)d_in[2];
    float*       out         = (float*)d_out;

    (void)in_sizes; (void)n_in; (void)d_ws; (void)ws_size; (void)out_size;

    zero_kernel<<<(OUT_TOT + 255) / 256, 256, 0, stream>>>(out, OUT_TOT);
    count_wmma_kernel<<<1, 128, 0, stream>>>(sel, out);
    rank_scatter_kernel<<<1, 1024, 0, stream>>>(pred_boxes, pred_scores, sel,
                                                out);
}
